// TransformerLayer_61426622267523
// MI455X (gfx1250) — compile-verified, matrix operand order checked by reading, not run on hardware
//
#include <hip/hip_runtime.h>
#include <stdint.h>

// ---------------------------------------------------------------------------
// Problem constants (match reference)
// ---------------------------------------------------------------------------
#define Hh    8
#define Dd    128
#define HIDD  128
#define Tt    64
#define Bb    64
#define KKk   704           // (N+1)*T
#define KHh   1024          // H*HID
#define BT    (Bb*Tt)       // 4096
#define HB    (Hh*Bb)       // 512

typedef _Float16 v16h __attribute__((ext_vector_type(16)));
typedef _Float16 v8h  __attribute__((ext_vector_type(8)));
typedef _Float16 v4h  __attribute__((ext_vector_type(4)));
typedef float    v8f  __attribute__((ext_vector_type(8)));
typedef unsigned int u32x4 __attribute__((ext_vector_type(4)));
typedef int          i32x8 __attribute__((ext_vector_type(8)));
typedef int          i32x4 __attribute__((ext_vector_type(4)));

#define VZERO8 {0.f,0.f,0.f,0.f,0.f,0.f,0.f,0.f}

#if __has_builtin(__builtin_amdgcn_tensor_load_to_lds)
#define HAVE_TDM 1
#else
#define HAVE_TDM 0
#endif

// ---------------------------------------------------------------------------
// 16x16x32 f16 WMMA fragment load (wave32, ISA 7.12.2 layout):
//   element j of v16h holds K = kbase + ((j>>3)<<4) + ((lane>>4)<<3) + (j&7)
// i.e. two contiguous 16-byte chunks per lane.  Works for both global and LDS
// source pointers (address-space inference emits global_load_b128 /
// ds_load_b128 respectively).
// ---------------------------------------------------------------------------
__device__ __forceinline__ v16h frag16(const _Float16* __restrict__ row, int kbase, int hi8) {
  v8h lo = *reinterpret_cast<const v8h*>(row + kbase + hi8);
  v8h hi = *reinterpret_cast<const v8h*>(row + kbase + 16 + hi8);
  return __builtin_shufflevector(lo, hi, 0,1,2,3,4,5,6,7,8,9,10,11,12,13,14,15);
}

__device__ __forceinline__ v8f wmma32(v16h a, v16h b, v8f c) {
  // D = A(16x32) x B(32x16) + C, f32 accumulate
  return __builtin_amdgcn_wmma_f32_16x16x32_f16(false, a, false, b, (short)0, c, false, false);
}

#if HAVE_TDM
// LDS byte offset of a __shared__ object (addrspace(3) pointers are 32-bit
// offsets from the LDS base).
__device__ __forceinline__ unsigned lds_off(const void* p) {
  return (unsigned)(unsigned long long)(__attribute__((address_space(3))) const void*)p;
}

// Tensor Data Mover: 2D tile load global -> LDS.
// D# per CDNA5 ISA ch.8: group0 = {count/lds_addr/global_addr/type},
// group1 = {data_size=8B, tensor_dim0=stride0=tile_dim0=row_units,
//           tensor_dim1=tile_dim1=nrows}, groups 2/3 zero (<=2D tensor).
// This toolchain exposes the 6-arg builtin:
//   (u32x4 g0, i32x8 g1, i32x4 g2, i32x4 g3, i32x8 g4, i32 cpol)
__device__ __forceinline__ void tdm_load_2d(unsigned lds_byte_off, const void* gptr,
                                            unsigned row_units, unsigned nrows) {
  unsigned long long ga = (unsigned long long)gptr;
  u32x4 g0;
  g0[0] = 1u;                                              // count=1 (valid), user mode
  g0[1] = lds_byte_off;                                    // lds_addr
  g0[2] = (unsigned)(ga & 0xffffffffu);                    // global_addr[31:0]
  g0[3] = (unsigned)((ga >> 32) & 0x01ffffffu) | (2u << 30); // global_addr[56:32], type=2
  i32x8 g1;
  g1[0] = (int)(3u << 16);                                 // data_size=3 (8B units)
  g1[1] = (int)((row_units & 0xffffu) << 16);              // tensor_dim0[15:0]
  g1[2] = (int)(((row_units >> 16) & 0xffffu) | ((nrows & 0xffffu) << 16)); // dim0[31:16], dim1[15:0]
  g1[3] = (int)(((nrows >> 16) & 0xffffu) | ((row_units & 0xffffu) << 16)); // dim1[31:16], tile_dim0
  g1[4] = (int)(nrows & 0xffffu);                          // tile_dim1 (tile_dim2=0)
  g1[5] = (int)row_units;                                  // tensor_dim0_stride[31:0]
  g1[6] = 0;                                               // stride0[47:32], dim1_stride[15:0]
  g1[7] = 0;
  i32x4 z4 = {0, 0, 0, 0};
  i32x8 z8 = {0, 0, 0, 0, 0, 0, 0, 0};
  __builtin_amdgcn_tensor_load_to_lds(g0, g1, z4, z4, z8, 0);
}
#endif

// Cooperative fallback / helper: plain block copy global f16 -> LDS.
__device__ __forceinline__ void coop_copy_lds(_Float16* dst, const _Float16* src,
                                              int halves, int tid, int nthreads) {
  for (int i = tid; i < halves / 8; i += nthreads)
    reinterpret_cast<v8h*>(dst)[i] = reinterpret_cast<const v8h*>(src)[i];
}

// ---------------------------------------------------------------------------
// K0: f32 -> f16 conversions + layout prep
// ---------------------------------------------------------------------------
__global__ void prep_kernel(const float* __restrict__ node, const float* __restrict__ neigh,
                            const float* __restrict__ wqs, const float* __restrict__ wks,
                            const float* __restrict__ wvs, const float* __restrict__ prjw,
                            _Float16* __restrict__ kv, _Float16* __restrict__ wqT,
                            _Float16* __restrict__ wkT, _Float16* __restrict__ wvT,
                            _Float16* __restrict__ pw) {
  const int KV_E = Bb * KKk * Dd;                   // 5,767,168
  const int TOTAL = KV_E + 4 * (1 << 17);
  int i = blockIdx.x * blockDim.x + threadIdx.x;
  if (i >= TOTAL) return;
  if (i < KV_E) {
    int d    = i & (Dd - 1);
    int rest = i >> 7;
    int kk   = rest % KKk;
    int b    = rest / KKk;
    int j = kk >> 6, t = kk & 63;
    float v = (j == 0) ? node[(b * Tt + t) * Dd + d]
                       : neigh[((b * 10 + (j - 1)) * Tt + t) * Dd + d];
    kv[i] = (_Float16)v;
  } else {
    int i2  = i - KV_E;
    int sel = i2 >> 17;
    int r   = i2 & 131071;
    if (sel < 3) {
      int h = r >> 14;
      int rr = r & 16383;
      int e = rr >> 7, d = rr & 127;
      const float* src = (sel == 0) ? wqs : (sel == 1) ? wks : wvs;
      _Float16* dst    = (sel == 0) ? wqT : (sel == 1) ? wkT : wvT;
      dst[r] = (_Float16)src[h * 16384 + d * HIDD + e];
    } else {
      pw[r] = (_Float16)prjw[r];
    }
  }
}

// ---------------------------------------------------------------------------
// K1: Q/K/V projections. One wave per 16x16 output tile, K-loop over D=128.
// ---------------------------------------------------------------------------
__global__ void __launch_bounds__(256) qkv_kernel(
    const _Float16* __restrict__ kv, const _Float16* __restrict__ wqT,
    const _Float16* __restrict__ wkT, const _Float16* __restrict__ wvT,
    _Float16* __restrict__ q, _Float16* __restrict__ k, _Float16* __restrict__ vt) {
  const int NQT = HB * 32;
  const int NKT = HB * 352;
  int wave = blockIdx.x * (blockDim.x >> 5) + (threadIdx.x >> 5);
  if (wave >= NQT + 2 * NKT) return;
  int lane = threadIdx.x & 31;
  int lr   = lane & 15;
  int hi8  = (lane >> 4) << 3;

  int mode, hb, mt, nt;
  if (wave < NQT)            { mode = 0; hb = wave >> 5; int tt = wave & 31;            mt = tt >> 3; nt = tt & 7; }
  else if (wave < NQT + NKT) { mode = 1; int w2 = wave - NQT;       hb = w2 / 352; int tt = w2 % 352; mt = tt >> 3; nt = tt & 7; }
  else                       { mode = 2; int w2 = wave - NQT - NKT; hb = w2 / 352; int tt = w2 % 352; mt = tt >> 3; nt = tt & 7; }
  int b = hb & (Bb - 1), h = hb >> 6;

  const _Float16* arow = kv + (size_t)(b * KKk + mt * 16 + lr) * Dd;
  const _Float16* wT   = (mode == 0) ? wqT : (mode == 1) ? wkT : wvT;
  const _Float16* brow = wT + (size_t)(h * HIDD + nt * 16 + lr) * Dd;

  v8f acc = VZERO8;
#pragma unroll
  for (int kb = 0; kb < Dd; kb += 32) {
    __builtin_prefetch(arow + kb + 64, 0, 1);
    acc = wmma32(frag16(arow, kb, hi8), frag16(brow, kb, hi8), acc);
  }

  if (mode == 0) {
#pragma unroll
    for (int vr = 0; vr < 8; ++vr) {
      int row = mt * 16 + vr + hi8;
      q[(size_t)(hb * Tt + row) * HIDD + nt * 16 + lr] = (_Float16)acc[vr];
    }
  } else if (mode == 1) {
#pragma unroll
    for (int vr = 0; vr < 8; ++vr) {
      int row = mt * 16 + vr + hi8;
      k[(size_t)(hb * KKk + row) * HIDD + nt * 16 + lr] = (_Float16)acc[vr];
    }
  } else {
#pragma unroll
    for (int vr = 0; vr < 8; ++vr) {
      int row = mt * 16 + vr + hi8;
      vt[(size_t)(hb * HIDD + nt * 16 + lr) * KKk + row] = (_Float16)acc[vr];
    }
  }
}

// ---------------------------------------------------------------------------
// K2: attention for one (h,b) per block; 8 waves.
//   Q tile (64x128 f16, 16KB) staged via Tensor Data Mover into LDS; each Q
//   fragment is re-read 22x in pass1 -> LDS hits instead of L2.
//   pass1 S=QK^T (WMMA) + mask + rowmax; pass2 exp/rowsum; pass3
//   O=P V^T (WMMA, A-fragment reuse 4x), scaled by 1/l.
// ---------------------------------------------------------------------------
__global__ void __launch_bounds__(256) attn_kernel(
    const _Float16* __restrict__ q, const _Float16* __restrict__ k,
    const _Float16* __restrict__ vt, const uint8_t* __restrict__ mask,
    float* __restrict__ sbuf, _Float16* __restrict__ p,
    float* __restrict__ invl, _Float16* __restrict__ o) {
  int hb = blockIdx.x;
  int b = hb & (Bb - 1), h = hb >> 6;
  int tid = threadIdx.x;
  int w = tid >> 5, lane = tid & 31, lr = lane & 15, hi8 = (lane >> 4) << 3;
  int rstrip = w & 3, chalf = w >> 2;

  __shared__ __align__(16) _Float16 qs[Tt * HIDD];   // 16 KB staged Q tile
  __shared__ float pmax[64][2];
  __shared__ float rowm[64];
  __shared__ float rsum[64];
  __shared__ float rinv[64];

  // ---- stage Q[hb] into LDS (TDM; one issue per block, TDM ignores EXEC) ----
#if HAVE_TDM
  if (tid == 0) {
    tdm_load_2d(lds_off(qs), q + (size_t)hb * Tt * HIDD,
                /*row 8B units*/ HIDD * 2 / 8, /*rows*/ Tt);
    __builtin_amdgcn_s_wait_tensorcnt(0);
  }
#else
  coop_copy_lds(qs, q + (size_t)hb * Tt * HIDD, Tt * HIDD, tid, 256);
#endif
  __syncthreads();

  // ---- pass 1: scores ----
  const _Float16* arow = qs + (rstrip * 16 + lr) * HIDD;
  float vmax[8];
#pragma unroll
  for (int vr = 0; vr < 8; ++vr) vmax[vr] = -__builtin_inff();

  for (int j = 0; j < 22; ++j) {
    int ct = chalf * 22 + j;
    const _Float16* brow = k + (size_t)(hb * KKk + ct * 16 + lr) * HIDD;
    v8f acc = VZERO8;
#pragma unroll
    for (int kb = 0; kb < HIDD; kb += 32)
      acc = wmma32(frag16(arow, kb, hi8), frag16(brow, kb, hi8), acc);
    int colk = ct * 16 + lr;
#pragma unroll
    for (int vr = 0; vr < 8; ++vr) {
      int trow = rstrip * 16 + vr + hi8;
      float sv = acc[vr];
      if (mask[(size_t)(b * Tt + trow) * KKk + colk]) sv = -__builtin_inff();
      sbuf[(size_t)(hb * Tt + trow) * KKk + colk] = sv;
      vmax[vr] = fmaxf(vmax[vr], sv);
    }
  }
#pragma unroll
  for (int vr = 0; vr < 8; ++vr) {
    float m = vmax[vr];
    for (int off = 1; off < 16; off <<= 1) m = fmaxf(m, __shfl_xor(m, off, 32));
    if (lr == 0) pmax[rstrip * 16 + vr + hi8][chalf] = m;
  }
  __syncthreads();
  if (tid < 64) { rowm[tid] = fmaxf(pmax[tid][0], pmax[tid][1]); rsum[tid] = 0.f; }
  __syncthreads();

  // ---- pass 2: exp + row sums (each wave re-reads its own 16x352 region) ----
  for (int i = lane; i < 16 * 88; i += 32) {
    int r = i / 88, c4 = i % 88;
    int trow = rstrip * 16 + r;
    int col  = chalf * 352 + c4 * 4;
    const float4 sv = *reinterpret_cast<const float4*>(sbuf + (size_t)(hb * Tt + trow) * KKk + col);
    float m  = rowm[trow];
    float e0 = __expf(sv.x - m), e1 = __expf(sv.y - m);
    float e2 = __expf(sv.z - m), e3 = __expf(sv.w - m);
    v4h pv = { (_Float16)e0, (_Float16)e1, (_Float16)e2, (_Float16)e3 };
    *reinterpret_cast<v4h*>(p + (size_t)(hb * Tt + trow) * KKk + col) = pv;
    atomicAdd(&rsum[trow], e0 + e1 + e2 + e3);        // ds_add_f32
  }
  __syncthreads();
  if (tid < 64) {
    float il = 1.0f / rsum[tid];
    rinv[tid] = il;
    invl[hb * Tt + tid] = il;
  }
  __syncthreads();

  // ---- pass 3: O = P x V^T, 4 N-tiles per wave, A-fragment reused 4x ----
  const _Float16* prow  = p  + (size_t)(hb * Tt + rstrip * 16 + lr) * KKk;
  const _Float16* vbase = vt + (size_t)(hb * HIDD + chalf * 64 + lr) * KKk;
  v8f accs[4] = { VZERO8, VZERO8, VZERO8, VZERO8 };
  for (int kb = 0; kb < KKk; kb += 32) {
    v16h a = frag16(prow, kb, hi8);
#pragma unroll
    for (int j = 0; j < 4; ++j)
      accs[j] = wmma32(a, frag16(vbase + (size_t)j * 16 * KKk, kb, hi8), accs[j]);
  }
#pragma unroll
  for (int j = 0; j < 4; ++j)
#pragma unroll
    for (int vr = 0; vr < 8; ++vr) {
      int trow = rstrip * 16 + vr + hi8;
      float ov = accs[j][vr] * rinv[trow];
      int e = chalf * 64 + j * 16 + lr;
      o[(size_t)(b * Tt + trow) * KHh + h * HIDD + e] = (_Float16)ov;
    }
}

// ---------------------------------------------------------------------------
// K3: out = LN(o @ prj_w^T + prj_b + node).  Block = 16 rows of BT.
//   All 8 waves consume the SAME 16x1024 A-strip -> stage once via TDM into
//   LDS (32 KB) and feed WMMA A-fragments from ds_load_b128 (8x less L2
//   traffic).  Wave w computes N-tile w over K=1024, then LN per row.
// ---------------------------------------------------------------------------
__global__ void __launch_bounds__(256) proj_ln_kernel(
    const _Float16* __restrict__ o, const _Float16* __restrict__ pw,
    const float* __restrict__ node, const float* __restrict__ prjb,
    const float* __restrict__ lng, const float* __restrict__ lnb,
    float* __restrict__ out) {
  int m = blockIdx.x;
  int tid = threadIdx.x;
  int w = tid >> 5, lane = tid & 31, lr = lane & 15, hi8 = (lane >> 4) << 3;
  __shared__ __align__(16) _Float16 as[16 * KHh];    // 32 KB staged A strip
  __shared__ float sres[16][HIDD];

#if HAVE_TDM
  if (tid == 0) {
    tdm_load_2d(lds_off(as), o + (size_t)m * 16 * KHh,
                /*row 8B units*/ KHh * 2 / 8, /*rows*/ 16);
    __builtin_amdgcn_s_wait_tensorcnt(0);
  }
#else
  coop_copy_lds(as, o + (size_t)m * 16 * KHh, 16 * KHh, tid, 256);
#endif
  __syncthreads();

  const _Float16* arow = as + lr * KHh;
  const _Float16* brow = pw + (size_t)(w * 16 + lr) * KHh;
  v8f acc = VZERO8;
  for (int kb = 0; kb < KHh; kb += 32)
    acc = wmma32(frag16(arow, kb, hi8), frag16(brow, kb, hi8), acc);
#pragma unroll
  for (int vr = 0; vr < 8; ++vr)
    sres[vr + hi8][w * 16 + lr] = acc[vr];
  __syncthreads();

  if (tid < 16) {
    int bt = m * 16 + tid;
    const float* nrow = node + (size_t)bt * Dd;
    float mu = 0.f;
    for (int e = 0; e < HIDD; ++e) {
      float v = sres[tid][e] + prjb[e] + nrow[e];
      sres[tid][e] = v;
      mu += v;
    }
    mu *= (1.0f / HIDD);
    float var = 0.f;
    for (int e = 0; e < HIDD; ++e) { float d = sres[tid][e] - mu; var += d * d; }
    var *= (1.0f / HIDD);
    float rs = rsqrtf(var + 1e-6f);
    for (int e = 0; e < HIDD; ++e)
      out[(size_t)bt * HIDD + e] = (sres[tid][e] - mu) * rs * lng[e] + lnb[e];
  }
}

// ---------------------------------------------------------------------------
// K4: slf_attn[bt,k] = sum_h probs[h,bt,k] * invl[h,bt]
// ---------------------------------------------------------------------------
__global__ void slf_kernel(const _Float16* __restrict__ p, const float* __restrict__ invl,
                           float* __restrict__ slf) {
  int idx = blockIdx.x * blockDim.x + threadIdx.x;
  if (idx >= BT * KKk / 4) return;
  int base = idx * 4;
  int bt = base / KKk;
  int kk = base % KKk;
  float a0 = 0, a1 = 0, a2 = 0, a3 = 0;
#pragma unroll
  for (int h = 0; h < Hh; ++h) {
    int hbt = h * BT + bt;
    v4h pv = *reinterpret_cast<const v4h*>(p + (size_t)hbt * KKk + kk);
    float il = invl[hbt];
    a0 += (float)pv[0] * il; a1 += (float)pv[1] * il;
    a2 += (float)pv[2] * il; a3 += (float)pv[3] * il;
  }
  float4 r = { a0, a1, a2, a3 };
  *reinterpret_cast<float4*>(slf + (size_t)bt * KKk + kk) = r;
}

// ---------------------------------------------------------------------------
// Host launcher
// ---------------------------------------------------------------------------
extern "C" void kernel_launch(void* const* d_in, const int* in_sizes, int n_in,
                              void* d_out, int out_size, void* d_ws, size_t ws_size,
                              hipStream_t stream) {
  const float*   node  = (const float*)d_in[0];
  const float*   neigh = (const float*)d_in[1];
  const uint8_t* mask  = (const uint8_t*)d_in[2];
  const float*   wqs   = (const float*)d_in[3];
  const float*   wks   = (const float*)d_in[4];
  const float*   wvs   = (const float*)d_in[5];
  const float*   prjw  = (const float*)d_in[6];
  const float*   prjb  = (const float*)d_in[7];
  const float*   lng   = (const float*)d_in[8];
  const float*   lnb   = (const float*)d_in[9];

  float* out = (float*)d_out;                 // [B,T,HID]
  float* slf = out + (size_t)BT * HIDD;       // [B,T,K]

  char* base = (char*)d_ws;
  size_t off = 0;
  auto carve = [&](size_t bytes) -> char* {
    char* r = base + off;
    off = (off + bytes + 255) & ~(size_t)255;
    return r;
  };
  _Float16* kv   = (_Float16*)carve((size_t)Bb * KKk * Dd * 2);
  _Float16* wqT  = (_Float16*)carve((size_t)Hh * HIDD * Dd * 2);
  _Float16* wkT  = (_Float16*)carve((size_t)Hh * HIDD * Dd * 2);
  _Float16* wvT  = (_Float16*)carve((size_t)Hh * HIDD * Dd * 2);
  _Float16* pw   = (_Float16*)carve((size_t)HIDD * KHh * 2);
  _Float16* q16  = (_Float16*)carve((size_t)HB * Tt * HIDD * 2);
  _Float16* k16  = (_Float16*)carve((size_t)HB * KKk * HIDD * 2);
  _Float16* vt16 = (_Float16*)carve((size_t)HB * HIDD * KKk * 2);
  float*    sbuf = (float*)   carve((size_t)HB * Tt * KKk * 4);
  _Float16* p16  = (_Float16*)carve((size_t)HB * Tt * KKk * 2);
  float*    il   = (float*)   carve((size_t)HB * Tt * 4);
  _Float16* o16  = (_Float16*)carve((size_t)BT * KHh * 2);

  {
    int total = Bb * KKk * Dd + 4 * (1 << 17);
    int blocks = (total + 255) / 256;
    prep_kernel<<<blocks, 256, 0, stream>>>(node, neigh, wqs, wks, wvs, prjw,
                                            kv, wqT, wkT, wvT, pw);
  }
  {
    int waves = HB * 32 + 2 * HB * 352;
    int blocks = waves / 8;
    qkv_kernel<<<blocks, 256, 0, stream>>>(kv, wqT, wkT, wvT, q16, k16, vt16);
  }
  attn_kernel<<<HB, 256, 0, stream>>>(q16, k16, vt16, mask, sbuf, p16, il, o16);
  proj_ln_kernel<<<BT / 16, 256, 0, stream>>>(o16, pw, node, prjb, lng, lnb, out);
  slf_kernel<<<(BT * KKk / 4 + 255) / 256, 256, 0, stream>>>(p16, il, slf);
}
